// MultiLayerAttentionCombine_85375359910645
// MI455X (gfx1250) — matrix operand order repaired and checked by reading, not compiled, not run on hardware
//
#include <hip/hip_runtime.h>
#include <hip/hip_bf16.h>
#include <stddef.h>

// ---------------------------------------------------------------------------
// MI455X / gfx1250: wave32, WMMA 16x16x32 bf16 (f32 accumulate).
// GEMM K-loops use double-buffered async global->LDS (ASYNCcnt) when the
// toolchain exposes the gfx1250 builtins; otherwise a pipelined sync path.
// ---------------------------------------------------------------------------

typedef __attribute__((ext_vector_type(16))) __bf16          v16bf;
typedef __attribute__((ext_vector_type(16))) unsigned short  v16u;
typedef __attribute__((ext_vector_type(8)))  float           v8f;
typedef __attribute__((ext_vector_type(4)))  int             v4i;

#define Bn   64
#define Tn   400
#define En   512
#define Hn   8
#define Dn   64
#define Mrows (Bn * Tn)          // 25600

#if defined(__has_builtin)
#  if __has_builtin(__builtin_amdgcn_global_load_async_to_lds_b128) && \
      __has_builtin(__builtin_amdgcn_s_wait_asynccnt)
#    define HAS_ASYNC 1
#  endif
#endif
#ifndef HAS_ASYNC
#  define HAS_ASYNC 0
#endif

#if HAS_ASYNC
static __device__ __forceinline__ void async_ld_b128(const void* g, void* l) {
  // builtin signature (probed via diagnostics): (global v4i*, local v4i*, Imm, Imm)
  __builtin_amdgcn_global_load_async_to_lds_b128(
      (__attribute__((address_space(1))) v4i*)(g),
      (__attribute__((address_space(3))) v4i*)(l), 0, 0);
}
#endif

static __device__ __forceinline__ unsigned short f2bf(float f) {
  union { __bf16 b; unsigned short u; } c; c.b = (__bf16)f; return c.u;
}
static __device__ __forceinline__ float bf2f(unsigned short s) {
  union { unsigned short u; __bf16 b; } c; c.u = s; return (float)c.b;
}
static __device__ __forceinline__ v16bf as_bf(v16u u) {
  union { v16u u; v16bf b; } c; c.u = u; return c.b;
}
static __device__ __forceinline__ v8f wmma_bf16(v16bf a, v16bf b, v8f c) {
  return __builtin_amdgcn_wmma_f32_16x16x32_bf16(
      /*neg_a=*/false, a, /*neg_b=*/false, b,
      /*c_mod=*/(short)0, c, /*reuse_a=*/false, /*reuse_b=*/false);
}
static __device__ __forceinline__ float gelu_exact(float x) {
  return 0.5f * x * (1.0f + erff(x * 0.70710678118654752f));
}

union U16x8 { uint4 q; unsigned short u[8]; };

// iRPE piecewise bucket -> table gather, computed on the fly (no B*H*T*T buffer).
static __device__ __forceinline__ float rpe_bias(const float* __restrict__ ctb,
                                                 const float* __restrict__ tab,
                                                 int t, int s) {
  float dx = ctb[2 * t]     - ctb[2 * s];
  float dy = ctb[2 * t + 1] - ctb[2 * s + 1];
  float d  = truncf(sqrtf(dx * dx + dy * dy) * (1.0f / 12.0f)); // PIX_PER_PATCH=12
  float ra = fabsf(d);
  float sg = (d > 0.f) ? 1.f : ((d < 0.f) ? -1.f : 0.f);
  // ALPHA=12, BETA=24, GAMMA=96 -> (BETA-ALPHA)/log(GAMMA/ALPHA)=12/log(8)
  float li = sg * fminf(rintf(12.0f + __logf(fmaxf(ra, 1e-6f) * (1.0f / 12.0f))
                              * 5.7707801635558535f), 24.0f);
  int idx = (int)((ra <= 12.0f) ? d : li) + 24;
  idx = min(max(idx, 0), 48);
  return tab[idx];
}

// f32 -> bf16 bulk conversion (weights, one pass).
__global__ void k_cvt(const float* __restrict__ src, unsigned short* __restrict__ dst, int n) {
  const int i = blockIdx.x * 256 + threadIdx.x;
  if (i < n) dst[i] = f2bf(src[i]);
}

__global__ void k_init(const float* __restrict__ x, float* __restrict__ X,
                       unsigned short* __restrict__ XB, int n) {
  const int i = blockIdx.x * 256 + threadIdx.x;
  if (i < n) { const float v = x[i]; X[i] = v; XB[i] = f2bf(v); }
}

// ---------------------------------------------------------------------------
// GEMM: C(M,N) = act(A_bf16(M,K) @ W_bf16(K,N) + bias). Workgroup tile
// 64(M) x 128(N), 8 waves in 2x4; each wave owns a 32x32 block = 4 WMMA
// per K-step. Double-buffered LDS tiles with async global->LDS loads.
// ---------------------------------------------------------------------------
template <int ACT /*0 none,1 gelu*/, int OUTBF>
__global__ __launch_bounds__(256)
void k_gemm(const unsigned short* __restrict__ A, const unsigned short* __restrict__ W,
            const float* __restrict__ bias, void* __restrict__ Cout,
            int M, int N, int K) {
  __shared__ __attribute__((aligned(16))) unsigned short As[2][64][40];
  __shared__ __attribute__((aligned(16))) unsigned short Ws[2][32][136];

  const int tid  = threadIdx.x;
  const int lane = tid & 31;
  const int wv   = tid >> 5;
  const int waveM = wv & 1;       // 0..1 -> 32 rows each
  const int waveN = wv >> 1;      // 0..3 -> 32 cols each
  const int gm0 = blockIdx.y * 64, gn0 = blockIdx.x * 128;

  const bool hi = lane >= 16;
  const int  l  = lane & 15;

  v8f acc00 = {}, acc01 = {}, acc10 = {}, acc11 = {};

  const int arow = tid >> 2, acol = (tid & 3) * 8;    // A: 64 x 32 loader
  const int wrow = tid >> 3, wcol = (tid & 7) * 16;   // W: 32 x 128 loader

  const unsigned short* Ag = A + (size_t)(gm0 + arow) * K + acol;
  const unsigned short* Wg = W + (size_t)wrow * N + gn0 + wcol;

  auto issue = [&](int k0, int buf) {
#if HAS_ASYNC
    async_ld_b128(Ag + k0,                    &As[buf][arow][acol]);
    async_ld_b128(Wg + (size_t)k0 * N,        &Ws[buf][wrow][wcol]);
    async_ld_b128(Wg + (size_t)k0 * N + 8,    &Ws[buf][wrow][wcol + 8]);
#else
    *(uint4*)&As[buf][arow][acol]     = *(const uint4*)(Ag + k0);
    *(uint4*)&Ws[buf][wrow][wcol]     = *(const uint4*)(Wg + (size_t)k0 * N);
    *(uint4*)&Ws[buf][wrow][wcol + 8] = *(const uint4*)(Wg + (size_t)k0 * N + 8);
#endif
  };

  issue(0, 0);
  int buf = 0;
  for (int k0 = 0; k0 < K; k0 += 32) {
    if (k0 + 32 < K) {
      issue(k0 + 32, buf ^ 1);         // prefetch next tile into alternate buffer
#if HAS_ASYNC
      __builtin_amdgcn_s_wait_asynccnt(3);   // retire current batch only
#endif
    } else {
#if HAS_ASYNC
      __builtin_amdgcn_s_wait_asynccnt(0);
#endif
    }
    __syncthreads();

    v16u au0, au1, bu0, bu1;
    #pragma unroll
    for (int h = 0; h < 16; ++h) {
      const int ka = (h < 8 ? h : h + 8) + (hi ? 8 : 0);   // A 16x32 lane layout
      au0[h] = As[buf][waveM * 32 + l][ka];
      au1[h] = As[buf][waveM * 32 + 16 + l][ka];
      const int kb = (h < 8 ? h : h + 8) + (hi ? 8 : 0);   // B 32x16 lane layout
      bu0[h] = Ws[buf][kb][waveN * 32 + l];
      bu1[h] = Ws[buf][kb][waveN * 32 + 16 + l];
    }
    const v16bf a0 = as_bf(au0), a1 = as_bf(au1);
    const v16bf b0 = as_bf(bu0), b1 = as_bf(bu1);
    acc00 = wmma_bf16(a0, b0, acc00);
    acc01 = wmma_bf16(a0, b1, acc01);
    acc10 = wmma_bf16(a1, b0, acc10);
    acc11 = wmma_bf16(a1, b1, acc11);
    __syncthreads();
    buf ^= 1;
  }

  const int n0 = gn0 + waveN * 32 + l;
  const int n1 = n0 + 16;
  const float bv0 = bias ? bias[n0] : 0.f;
  const float bv1 = bias ? bias[n1] : 0.f;
  #pragma unroll
  for (int r = 0; r < 8; ++r) {
    const int m0 = gm0 + waveM * 32 + (hi ? 8 : 0) + r;   // C f32 lane layout
    const int m1 = m0 + 16;
    float v00 = acc00[r] + bv0, v01 = acc01[r] + bv1;
    float v10 = acc10[r] + bv0, v11 = acc11[r] + bv1;
    if (ACT == 1) {
      v00 = gelu_exact(v00); v01 = gelu_exact(v01);
      v10 = gelu_exact(v10); v11 = gelu_exact(v11);
    }
    if (OUTBF) {
      unsigned short* C = (unsigned short*)Cout;
      C[(size_t)m0 * N + n0] = f2bf(v00); C[(size_t)m0 * N + n1] = f2bf(v01);
      C[(size_t)m1 * N + n0] = f2bf(v10); C[(size_t)m1 * N + n1] = f2bf(v11);
    } else {
      float* C = (float*)Cout;
      C[(size_t)m0 * N + n0] = v00; C[(size_t)m0 * N + n1] = v01;
      C[(size_t)m1 * N + n0] = v10; C[(size_t)m1 * N + n1] = v11;
    }
  }
}

// ---------------------------------------------------------------------------
// Fused attention: one wave per (b, h, 16-query tile). S strip (16 x 416) in
// LDS, softmax in place (lane pairs split each row), P@V via WMMA with the
// V chunk staged row-major in LDS (b128/async loads, transpose in the read).
// ---------------------------------------------------------------------------
template <int RPE>
__global__ __launch_bounds__(32)
void k_attn(const unsigned short* __restrict__ QKV, const float* __restrict__ ct,
            const float* __restrict__ rpe_table, unsigned short* __restrict__ Y) {
  __shared__ float          sc[16][417];
  __shared__ __attribute__((aligned(16))) unsigned short Vl[32][64];

  const int b = blockIdx.z, h = blockIdx.y, t0 = blockIdx.x * 16;
  const int lane = threadIdx.x & 31;
  const bool hi = lane >= 16;
  const int  l  = lane & 15;

  // Q fragments (two K=32 chunks over D=64), loaded straight from global bf16.
  v16u qu[2];
  #pragma unroll
  for (int kc = 0; kc < 2; ++kc) {
    const unsigned short* qp =
        QKV + (size_t)(b * Tn + t0 + l) * 1536 + h * 64 + kc * 32 + (hi ? 8 : 0);
    U16x8 c0 = *(const U16x8*)qp;
    U16x8 c1 = *(const U16x8*)(qp + 16);
    #pragma unroll
    for (int j = 0; j < 8; ++j) { qu[kc][j] = c0.u[j]; qu[kc][8 + j] = c1.u[j]; }
  }

  const float* ctb = ct + b * (Tn * 2);
  const float* th  = RPE ? (rpe_table + h * 49) : nullptr;

  // ---- S = Q K^T / 8 (+ rpe), strip into LDS ----
  for (int st = 0; st < 25; ++st) {
    const int s0 = st * 16;
    v8f c = {};
    #pragma unroll
    for (int kc = 0; kc < 2; ++kc) {
      const unsigned short* kp = QKV + (size_t)(b * Tn + s0 + l) * 1536 + 512 +
                                 h * 64 + kc * 32 + (hi ? 8 : 0);
      U16x8 c0 = *(const U16x8*)kp;
      U16x8 c1 = *(const U16x8*)(kp + 16);
      v16u bu;
      #pragma unroll
      for (int j = 0; j < 8; ++j) { bu[j] = c0.u[j]; bu[8 + j] = c1.u[j]; }
      c = wmma_bf16(as_bf(qu[kc]), as_bf(bu), c);
    }
    #pragma unroll
    for (int r = 0; r < 8; ++r) {
      const int m = (hi ? 8 : 0) + r;
      float v = c[r] * 0.125f;                 // 1/sqrt(64)
      if (RPE) v += rpe_bias(ctb, th, t0 + m, s0 + l);
      sc[m][s0 + l] = v;
    }
  }
  __syncthreads();

  // ---- softmax: row l handled by lane pair (l, l+16), 200 cols each ----
  {
    const int off = hi ? 200 : 0;
    float mx = -1e30f;
    for (int s = 0; s < 200; ++s) mx = fmaxf(mx, sc[l][off + s]);
    mx = fmaxf(mx, __shfl_xor(mx, 16));
    float sum = 0.f;
    for (int s = 0; s < 200; ++s) {
      const float e = __expf(sc[l][off + s] - mx);
      sc[l][off + s] = e;
      sum += e;
    }
    sum += __shfl_xor(sum, 16);
    const float inv = __builtin_amdgcn_rcpf(sum);
    for (int s = 0; s < 200; ++s) sc[l][off + s] *= inv;
    if (hi) {
      for (int s = Tn; s < 416; ++s) sc[l][s] = 0.f;   // pad -> P=0
    }
  }
  __syncthreads();

  // ---- Y = P V ----
  v8f acc[4];
  #pragma unroll
  for (int i = 0; i < 4; ++i)
    #pragma unroll
    for (int r = 0; r < 8; ++r) acc[i][r] = 0.f;

  for (int sc32 = 0; sc32 < 13; ++sc32) {       // 13 * 32 = 416
    const int sbase = sc32 * 32;
    // V chunk (32 rows x 64 cols) staged row-major; OOB rows clamp to a valid
    // (finite) row -- their P weights are exactly zero.
    #pragma unroll
    for (int q8 = 0; q8 < 8; ++q8) {
      const int idx8 = lane + q8 * 32;          // 256 x b128 chunks
      const int sl = idx8 >> 3, dc = (idx8 & 7) * 8;
      const int s = min(sbase + sl, Tn - 1);
      const unsigned short* vp = QKV + (size_t)(b * Tn + s) * 1536 + 1024 + h * 64 + dc;
#if HAS_ASYNC
      async_ld_b128(vp, &Vl[sl][dc]);
#else
      *(uint4*)&Vl[sl][dc] = *(const uint4*)vp;
#endif
    }
#if HAS_ASYNC
    __builtin_amdgcn_s_wait_asynccnt(0);
#endif
    __syncthreads();

    v16bf pa;
    #pragma unroll
    for (int hh = 0; hh < 16; ++hh) {
      const int k = (hh < 8 ? hh : hh + 8) + (hi ? 8 : 0);
      pa[hh] = (__bf16)sc[l][sbase + k];
    }
    #pragma unroll
    for (int dt = 0; dt < 4; ++dt) {
      v16u vu;
      #pragma unroll
      for (int hh = 0; hh < 16; ++hh) vu[hh] = Vl[(hh < 8 ? hh : hh + 8) + (hi ? 8 : 0)][dt * 16 + l];
      acc[dt] = wmma_bf16(pa, as_bf(vu), acc[dt]);
    }
    __syncthreads();
  }

  #pragma unroll
  for (int dt = 0; dt < 4; ++dt)
    #pragma unroll
    for (int r = 0; r < 8; ++r) {
      const int t = t0 + (hi ? 8 : 0) + r;
      Y[(size_t)(b * Tn + t) * En + h * 64 + dt * 16 + l] = f2bf(acc[dt][r]);
    }
}

// ---------------------------------------------------------------------------
// Residual add + LayerNorm over E=512; writes f32 state and bf16 mirror.
// ---------------------------------------------------------------------------
__global__ __launch_bounds__(256)
void k_add_ln(float* __restrict__ X, const float* __restrict__ R,
              const float* __restrict__ g, const float* __restrict__ bta,
              unsigned short* __restrict__ XB) {
  __shared__ float red[2][8];
  const int row = blockIdx.x, tid = threadIdx.x;
  const size_t base = (size_t)row * En;
  float v0 = X[base + tid]       + R[base + tid];
  float v1 = X[base + 256 + tid] + R[base + 256 + tid];
  float s = v0 + v1, sq = v0 * v0 + v1 * v1;
  #pragma unroll
  for (int o = 16; o > 0; o >>= 1) { s += __shfl_xor(s, o); sq += __shfl_xor(sq, o); }
  const int wv = tid >> 5, ln = tid & 31;
  if (!ln) { red[0][wv] = s; red[1][wv] = sq; }
  __syncthreads();
  float S = 0.f, SQ = 0.f;
  #pragma unroll
  for (int i = 0; i < 8; ++i) { S += red[0][i]; SQ += red[1][i]; }
  const float mean = S * (1.0f / En);
  const float var  = SQ * (1.0f / En) - mean * mean;
  const float inv  = rsqrtf(var + 1e-5f);
  const float y0 = (v0 - mean) * inv * g[tid]       + bta[tid];
  const float y1 = (v1 - mean) * inv * g[256 + tid] + bta[256 + tid];
  X[base + tid] = y0;         X[base + 256 + tid] = y1;
  XB[base + tid] = f2bf(y0);  XB[base + 256 + tid] = f2bf(y1);
}

// ---------------------------------------------------------------------------
// pred_attention: att[b,t,s] = sigmoid( (1/8) * sum_e q[e]*wp[e/64]*k[e] ).
// Per-head Conv1d weights folded into a per-column scale on Q -> single
// K=512 WMMA chain per 16x16 output tile.
// ---------------------------------------------------------------------------
__global__ __launch_bounds__(32)
void k_pred(const unsigned short* __restrict__ QK, const float* __restrict__ wp,
            float* __restrict__ out) {
  const int b = blockIdx.z, s0 = blockIdx.y * 16, t0 = blockIdx.x * 16;
  const int lane = threadIdx.x & 31;
  const bool hi = lane >= 16;
  const int  l  = lane & 15;

  float w[8];
  #pragma unroll
  for (int i = 0; i < 8; ++i) w[i] = wp[i];

  v8f c = {};
  for (int kc = 0; kc < 16; ++kc) {
    v16bf pa;
    const unsigned short* qp = QK + (size_t)(b * Tn + t0 + l) * 1024 + kc * 32;
    #pragma unroll
    for (int hh = 0; hh < 16; ++hh) {
      const int k = (hh < 8 ? hh : hh + 8) + (hi ? 8 : 0);
      const int e = kc * 32 + k;
      pa[hh] = (__bf16)(bf2f(qp[k]) * w[e >> 6]);
    }
    const unsigned short* kp = QK + (size_t)(b * Tn + s0 + l) * 1024 + 512 +
                               kc * 32 + (hi ? 8 : 0);
    U16x8 c0 = *(const U16x8*)kp;
    U16x8 c1 = *(const U16x8*)(kp + 16);
    v16u bu;
    #pragma unroll
    for (int j = 0; j < 8; ++j) { bu[j] = c0.u[j]; bu[8 + j] = c1.u[j]; }
    c = wmma_bf16(pa, as_bf(bu), c);
  }
  #pragma unroll
  for (int r = 0; r < 8; ++r) {
    const int t = t0 + (hi ? 8 : 0) + r;
    const float v = c[r] * 0.125f;
    float sg = __builtin_amdgcn_rcpf(1.f + __expf(-v));
    if (sg != sg) sg = 0.f;                     // NaN guard (reference)
    out[((size_t)b * Tn + t) * Tn + s0 + l] = sg;
  }
}

// ---------------------------------------------------------------------------
// Host launcher. Params flattened jax-pytree style (sorted dict keys).
// ---------------------------------------------------------------------------
struct BP {
  const float *b_attn, *b_fc, *b_fc2, *b_proj;
  const float *ln1_b, *ln1_g, *ln2_b, *ln2_g;
  const float *rpe, *w_attn, *w_fc, *w_fc2, *w_proj;
};

extern "C" void kernel_launch(void* const* d_in, const int* in_sizes, int n_in,
                              void* d_out, int out_size, void* d_ws, size_t ws_size,
                              hipStream_t stream) {
  (void)in_sizes; (void)n_in; (void)out_size; (void)ws_size;
  const float* x  = (const float*)d_in[0];
  const float* ct = (const float*)d_in[1];

  int idx = 2;
  BP bp[3];
  for (int L = 0; L < 3; ++L) {
    bp[L].b_attn = (const float*)d_in[idx++];
    bp[L].b_fc   = (const float*)d_in[idx++];
    bp[L].b_fc2  = (const float*)d_in[idx++];
    bp[L].b_proj = (const float*)d_in[idx++];
    bp[L].ln1_b  = (const float*)d_in[idx++];
    bp[L].ln1_g  = (const float*)d_in[idx++];
    bp[L].ln2_b  = (const float*)d_in[idx++];
    bp[L].ln2_g  = (const float*)d_in[idx++];
    bp[L].rpe    = (L == 0) ? (const float*)d_in[idx++] : nullptr;
    bp[L].w_attn = (const float*)d_in[idx++];
    bp[L].w_fc   = (const float*)d_in[idx++];
    bp[L].w_fc2  = (const float*)d_in[idx++];
    bp[L].w_proj = (const float*)d_in[idx++];
  }
  const float* pred_b  = (const float*)d_in[idx++];
  const float* pred_w  = (const float*)d_in[idx++];
  const float* pred_wp = (const float*)d_in[idx++];

  // Workspace layout (bytes), all 256-aligned.
  char* ws = (char*)d_ws;
  float*          X32 = (float*)(ws + 0);                    // 52,428,800
  unsigned short* XB  = (unsigned short*)(ws + 52428800);    // 26,214,400
  unsigned short* QKV = (unsigned short*)(ws + 78643200);    // 78,643,200
  unsigned short* YB  = (unsigned short*)(ws + 157286400);   // 26,214,400
  unsigned short* HID = (unsigned short*)(ws + 183500800);   // 52,428,800
  float*          TMP = (float*)(ws + 235929600);            // 52,428,800
  unsigned short* WB  = (unsigned short*)(ws + 288358400);   // 13,631,488 (bf16 weights)

  // bf16 weight mirrors (one conversion pass; GEMMs then stream pure bf16).
  const int SZ_ATTN = En * 1536, SZ_FC = En * 1024, SZ_FC2 = 1024 * En, SZ_PROJ = En * En;
  unsigned short* wb_attn[3]; unsigned short* wb_fc[3];
  unsigned short* wb_fc2[3];  unsigned short* wb_proj[3];
  size_t o = 0;
  for (int L = 0; L < 3; ++L) {
    wb_attn[L] = WB + o; o += SZ_ATTN;
    wb_fc[L]   = WB + o; o += SZ_FC;
    wb_fc2[L]  = WB + o; o += SZ_FC2;
    wb_proj[L] = WB + o; o += SZ_PROJ;
  }
  unsigned short* wb_pred = WB + o;   // 512*1024

  for (int L = 0; L < 3; ++L) {
    k_cvt<<<dim3((SZ_ATTN + 255) / 256), 256, 0, stream>>>(bp[L].w_attn, wb_attn[L], SZ_ATTN);
    k_cvt<<<dim3((SZ_FC   + 255) / 256), 256, 0, stream>>>(bp[L].w_fc,   wb_fc[L],   SZ_FC);
    k_cvt<<<dim3((SZ_FC2  + 255) / 256), 256, 0, stream>>>(bp[L].w_fc2,  wb_fc2[L],  SZ_FC2);
    k_cvt<<<dim3((SZ_PROJ + 255) / 256), 256, 0, stream>>>(bp[L].w_proj, wb_proj[L], SZ_PROJ);
  }
  k_cvt<<<dim3((SZ_FC + 255) / 256), 256, 0, stream>>>(pred_w, wb_pred, SZ_FC);

  const int NE = Mrows * En;  // 13,107,200
  k_init<<<dim3((NE + 255) / 256), 256, 0, stream>>>(x, X32, XB, NE);

  for (int L = 0; L < 3; ++L) {
    // QKV = X @ w_attn + b_attn  (bf16 out)
    k_gemm<0, 1><<<dim3(1536 / 128, Mrows / 64), 256, 0, stream>>>(
        XB, wb_attn[L], bp[L].b_attn, QKV, Mrows, 1536, En);
    // fused attention
    if (L == 0)
      k_attn<1><<<dim3(25, Hn, Bn), 32, 0, stream>>>(QKV, ct, bp[0].rpe, YB);
    else
      k_attn<0><<<dim3(25, Hn, Bn), 32, 0, stream>>>(QKV, ct, nullptr, YB);
    // proj (f32 out)
    k_gemm<0, 0><<<dim3(En / 128, Mrows / 64), 256, 0, stream>>>(
        YB, wb_proj[L], bp[L].b_proj, TMP, Mrows, En, En);
    // X = LN(X + proj)
    k_add_ln<<<dim3(Mrows), 256, 0, stream>>>(X32, TMP, bp[L].ln1_g, bp[L].ln1_b, XB);
    // HID = gelu(X @ w_fc + b_fc)  (bf16 out)
    k_gemm<1, 1><<<dim3(1024 / 128, Mrows / 64), 256, 0, stream>>>(
        XB, wb_fc[L], bp[L].b_fc, HID, Mrows, 1024, En);
    // TMP = HID @ w_fc2 + b_fc2  (f32 out)
    k_gemm<0, 0><<<dim3(En / 128, Mrows / 64), 256, 0, stream>>>(
        HID, wb_fc2[L], bp[L].b_fc2, TMP, Mrows, En, 1024);
    // X = LN(X + mlp)
    k_add_ln<<<dim3(Mrows), 256, 0, stream>>>(X32, TMP, bp[L].ln2_g, bp[L].ln2_b, XB);
  }

  // pred: QK = X @ pred_w_attn + pred_b_attn  (bf16 out, reuse QKV buffer)
  k_gemm<0, 1><<<dim3(1024 / 128, Mrows / 64), 256, 0, stream>>>(
      XB, wb_pred, pred_b, QKV, Mrows, 1024, En);
  k_pred<<<dim3(25, 25, Bn), 32, 0, stream>>>(QKV, pred_wp, (float*)d_out);
}
